// ChimeraSurrogateNCA_76965813944637
// MI455X (gfx1250) — matrix operand order repaired and checked by reading, not compiled, not run on hardware
//
#include <hip/hip_runtime.h>
#include <cmath>

typedef __attribute__((ext_vector_type(16))) __bf16 v16bf;
typedef __attribute__((ext_vector_type(8)))  __bf16 v8bf;
typedef __attribute__((ext_vector_type(8)))  float  v8f;
typedef __attribute__((ext_vector_type(8)))  unsigned int v8u;

#define C_DIM 256
#define HDIM  128
#define WDIM  128
#define HW    (HDIM*WDIM)
#define KC    32          // input channels staged per K-chunk
#define THT   8           // pixel tile height
#define TWT   16          // pixel tile width  (128 pixels per workgroup)
#define HH    10          // halo height
#define HWID  18          // halo width
#define BUFSZ (HH*HWID*KC) // 5760 bf16 = 11.5 KB per buffer (double buffered)

// ---------------------------------------------------------------------------
// Repack fp32 weights [o][c][ky][kx] -> bf16 [k][o][c] so A-fragments for
// V_WMMA_F32_16X16X32_BF16 are two 16B per-lane loads in contiguous k order (k = 16h + s).
// ---------------------------------------------------------------------------
__global__ __launch_bounds__(256) void repack_weights(const float* __restrict__ w,
                                                      __bf16* __restrict__ wk) {
    int idx = blockIdx.x * 256 + threadIdx.x;
    if (idx >= 9 * C_DIM * C_DIM) return;
    int k = idx % 9;
    int t = idx / 9;
    int c = t % C_DIM;
    int o = t / C_DIM;
    wk[(k * C_DIM + o) * C_DIM + c] = (__bf16)w[idx];
}

// ---------------------------------------------------------------------------
// One NCA step: xout = xin + f(conv_masked(xin)), f(d) = d/(1+|d|).
// Grid: (128 pixel tiles, 2 M-blocks of 128 out-ch, 4 batches). Block: 256.
// Wave tile: 32 out-channels x 64 pixels (2 A-frags x 4 B-frags, 8 WMMA/tap).
// ---------------------------------------------------------------------------
__global__ __launch_bounds__(256) void nca_step(const float* __restrict__ xin,
                                                float* __restrict__ xout,
                                                const __bf16* __restrict__ wk,
                                                const float* __restrict__ mask) {
    __shared__ __align__(128) __bf16 lx[2 * BUFSZ];

    const int tid   = threadIdx.x;
    const int lane  = tid & 31;
    const int wave  = tid >> 5;
    const int mw    = wave >> 1;   // 0..3 : 32-channel M strip
    const int nw    = wave & 1;    // 0..1 : 64-pixel N strip
    const int lhalf = lane >> 4;   // 0 / 1
    const int l15   = lane & 15;

    const int bt     = blockIdx.x;          // 0..127 pixel tile
    const int mblock = blockIdx.y;          // 0..1
    const int b      = blockIdx.z;          // 0..3

    const int h0    = (bt >> 3) * THT;      // 16 tile rows
    const int w0    = (bt & 7) * TWT;       // 8 tile cols
    const int mBase = mblock * 128 + mw * 32;

    // Per-lane pixel coords for 4 N-subtiles; masks packed 9 bits/subtile.
    int ghp[4], gwp[4], boff[4];
    unsigned mbits[4];
#pragma unroll
    for (int s = 0; s < 4; ++s) {
        int p  = nw * 64 + s * 16 + l15;
        int py = p >> 4, px = p & 15;
        ghp[s] = h0 + py;
        gwp[s] = w0 + px;
        boff[s] = (py * HWID + px) * KC + lhalf * 16;   // halo base (dy=dx=0)
        unsigned m = 0;
#pragma unroll
        for (int k = 0; k < 9; ++k)
            m |= (mask[k * HW + ghp[s] * WDIM + gwp[s]] != 0.0f) ? (1u << k) : 0u;
        mbits[s] = m;
    }

    // Per-lane A base pointers (tap & K-chunk fold into constant offsets).
    const __bf16* abase[2];
#pragma unroll
    for (int mi = 0; mi < 2; ++mi)
        abase[mi] = wk + (size_t)(mBase + mi * 16 + l15) * C_DIM + lhalf * 16;

    v8f zero = {0.f, 0.f, 0.f, 0.f, 0.f, 0.f, 0.f, 0.f};
    v8f acc[2][4];
#pragma unroll
    for (int mi = 0; mi < 2; ++mi)
#pragma unroll
        for (int si = 0; si < 4; ++si) acc[mi][si] = zero;

    // Cooperative halo stage of one 32-channel chunk into LDS buffer (kc&1).
    auto stage = [&](int kc) {
        const int c0 = kc * KC;
        __bf16* dst = lx + (kc & 1) * BUFSZ;
        for (int e = tid; e < HH * HWID * KC; e += 256) {
            int ci  = e / (HH * HWID);
            int pix = e - ci * (HH * HWID);
            int hy  = pix / HWID;
            int hx  = pix - hy * HWID;
            int gh  = h0 - 1 + hy;
            int gw  = w0 - 1 + hx;
            float v = 0.0f;
            if ((unsigned)gh < (unsigned)HDIM && (unsigned)gw < (unsigned)WDIM)
                v = xin[(((size_t)(b * C_DIM + c0 + ci)) << 14) + (gh << 7) + gw];
            dst[pix * KC + ci] = (__bf16)v;
        }
    };

    stage(0);
    __syncthreads();

    for (int kc = 0; kc < C_DIM / KC; ++kc) {
        if (kc < C_DIM / KC - 1) stage(kc + 1);   // prefetch next chunk
        const __bf16* bufp = lx + (kc & 1) * BUFSZ;
        const int c0 = kc * KC;

#pragma unroll
        for (int k = 0; k < 9; ++k) {
            const int dy = k / 3, dx = k % 3;

            // A fragments: constant per-tap offset folds into load offset.
            v16bf afr[2];
#pragma unroll
            for (int mi = 0; mi < 2; ++mi) {
                const __bf16* ap = abase[mi] + k * (C_DIM * C_DIM) + c0;
                v8bf a0 = *(const v8bf*)ap;
                v8bf a1 = *(const v8bf*)(ap + 8);
                afr[mi] = __builtin_shufflevector(a0, a1, 0, 1, 2, 3, 4, 5, 6, 7,
                                                  8, 9, 10, 11, 12, 13, 14, 15);
            }

#pragma unroll
            for (int s = 0; s < 4; ++s) {
                // B fragment: shifted LDS read (tap offset is an immediate),
                // masked branchlessly: sign-extend bit k of packed mask word.
                v16bf bv = *(const v16bf*)(bufp + boff[s] + (dy * HWID + dx) * KC);
                unsigned sel = (unsigned)((int)(mbits[s] << (31 - k)) >> 31);
                v8u bu = __builtin_bit_cast(v8u, bv);
                bu &= sel;
                v16bf bfr = __builtin_bit_cast(v16bf, bu);

                acc[0][s] = __builtin_amdgcn_wmma_f32_16x16x32_bf16(
                    false, afr[0], false, bfr, (short)0, acc[0][s], false, false);
                acc[1][s] = __builtin_amdgcn_wmma_f32_16x16x32_bf16(
                    false, afr[1], false, bfr, (short)0, acc[1][s], false, false);
            }
        }
        __syncthreads();
    }

    // Epilogue: x + d/(1+|d|). D layout: lane<16 -> M=r, lane>=16 -> M=8+r.
#pragma unroll
    for (int mi = 0; mi < 2; ++mi) {
#pragma unroll
        for (int si = 0; si < 4; ++si) {
#pragma unroll
            for (int r = 0; r < 8; ++r) {
                int ch = mBase + mi * 16 + lhalf * 8 + r;
                size_t idx = (((size_t)(b * C_DIM + ch)) << 14) +
                             (ghp[si] << 7) + gwp[si];
                float d  = acc[mi][si][r];
                float xv = xin[idx];
                xout[idx] = xv + d / (1.0f + fabsf(d));
            }
        }
    }
}

// ---------------------------------------------------------------------------
extern "C" void kernel_launch(void* const* d_in, const int* in_sizes, int n_in,
                              void* d_out, int out_size, void* d_ws, size_t ws_size,
                              hipStream_t stream) {
    (void)in_sizes; (void)n_in; (void)out_size; (void)ws_size;

    const float* retina = (const float*)d_in[0];
    const float* weight = (const float*)d_in[1];
    const float* mask   = (const float*)d_in[2];
    float* out = (float*)d_out;

    __bf16* wk  = (__bf16*)d_ws;                        // 1.125 MB bf16 weights
    float*  tmp = (float*)((char*)d_ws + (4ull << 20)); // 64 MB ping-pong state

    repack_weights<<<(9 * C_DIM * C_DIM + 255) / 256, 256, 0, stream>>>(weight, wk);

    dim3 grid(128, 2, 4), blk(256);
    // 5 steps, ping-pong so the final result lands in d_out.
    nca_step<<<grid, blk, 0, stream>>>(retina, out, wk, mask);
    nca_step<<<grid, blk, 0, stream>>>(out, tmp, wk, mask);
    nca_step<<<grid, blk, 0, stream>>>(tmp, out, wk, mask);
    nca_step<<<grid, blk, 0, stream>>>(out, tmp, wk, mask);
    nca_step<<<grid, blk, 0, stream>>>(tmp, out, wk, mask);
}
